// MultiHeadAttention_755914244305
// MI455X (gfx1250) — matrix operand order repaired and checked by reading, not compiled, not run on hardware
//
#include <hip/hip_runtime.h>
#include <hip/hip_bf16.h>

// MI455X / gfx1250 — wave32, WMMA bf16 path.
typedef __attribute__((ext_vector_type(16))) __bf16 v16bf;
typedef __attribute__((ext_vector_type(8)))  __bf16 v8bf;
typedef __attribute__((ext_vector_type(8)))  float  v8f;
typedef __attribute__((ext_vector_type(4)))  float  v4f;

#define B_  4
#define L_  1024
#define D_  1024
#define H_  16
#define DK_ 64

static __device__ __forceinline__ v8f wmma_bf16(v16bf a, v16bf b, v8f c) {
    // (neg_a, A, neg_b, B, c_mod, C, reuse_a, reuse_b)
    return __builtin_amdgcn_wmma_f32_16x16x32_bf16(false, a, false, b, (short)0, c,
                                                   false, false);
}

// ---------------------------------------------------------------------------
// f32 -> bf16 cast (grid-stride)
// ---------------------------------------------------------------------------
__global__ void cast_f32_bf16(const float* __restrict__ in, __bf16* __restrict__ out,
                              int n) {
    int i = blockIdx.x * blockDim.x + threadIdx.x;
    int stride = gridDim.x * blockDim.x;
    for (; i < n; i += stride) out[i] = (__bf16)in[i];
}

// ---------------------------------------------------------------------------
// QKV projection: O = X @ W^T  (X:[4096,1024] bf16, W:[1024,1024] bf16 row-major)
// z=0 -> qh [B,H,L,64], z=1 -> kh [B,H,L,64], z=2 -> vht [B,H,64,L] (transposed)
// Block: 256 threads = 8 waves; wave computes a 16(M) x 64(N) strip.
// K loop is software-pipelined (double-buffered fragments) so next-step loads
// are in flight while current WMMAs execute -> partial s_wait_loadcnt.
// ---------------------------------------------------------------------------
__global__ void proj_kernel(const __bf16* __restrict__ qb, const __bf16* __restrict__ kb,
                            const __bf16* __restrict__ vb,
                            const __bf16* __restrict__ wq, const __bf16* __restrict__ wk,
                            const __bf16* __restrict__ wv,
                            __bf16* __restrict__ qh, __bf16* __restrict__ kh,
                            __bf16* __restrict__ vht) {
    const int z = blockIdx.z;
    const __bf16* X = (z == 0) ? qb : (z == 1) ? kb : vb;
    const __bf16* W = (z == 0) ? wq : (z == 1) ? wk : wv;
    __bf16*       O = (z == 0) ? qh : (z == 1) ? kh : vht;

    const int wave = threadIdx.x >> 5;
    const int lane = threadIdx.x & 31;
    const int hf   = lane >> 4;      // half-wave select
    const int mn   = lane & 15;      // row (A) / col (B) within tile
    const int m_tile = blockIdx.y * 128 + wave * 16;
    const int n_base = blockIdx.x * 64;

    const __bf16* Arow = X + (m_tile + mn) * D_;
    const __bf16* Bcol = W + (n_base + mn) * D_ + 16 * hf;  // + t*16*D_ per tile

    v8f acc[4] = {};

    // ---- prologue: fragments for kk = 0
    v8bf alo = *reinterpret_cast<const v8bf*>(Arow + 16 * hf);
    v8bf ahi = *reinterpret_cast<const v8bf*>(Arow + 8 + 16 * hf);
    v16bf bcur[4];
#pragma unroll
    for (int t = 0; t < 4; ++t)
        bcur[t] = *reinterpret_cast<const v16bf*>(Bcol + t * 16 * D_);

    for (int kk = 0; kk < D_ - 32; kk += 32) {
        const int kn = kk + 32;
        // prefetch next K-step fragments
        v8bf alo_n = *reinterpret_cast<const v8bf*>(Arow + kn + 16 * hf);
        v8bf ahi_n = *reinterpret_cast<const v8bf*>(Arow + kn + 8 + 16 * hf);
        v16bf bnxt[4];
#pragma unroll
        for (int t = 0; t < 4; ++t)
            bnxt[t] = *reinterpret_cast<const v16bf*>(Bcol + t * 16 * D_ + kn);

        // compute current
        v16bf a;
#pragma unroll
        for (int e = 0; e < 8; ++e) { a[e] = alo[e]; a[8 + e] = ahi[e]; }
#pragma unroll
        for (int t = 0; t < 4; ++t) acc[t] = wmma_bf16(a, bcur[t], acc[t]);

        alo = alo_n; ahi = ahi_n;
#pragma unroll
        for (int t = 0; t < 4; ++t) bcur[t] = bnxt[t];
    }
    // ---- epilogue: last K-step
    {
        v16bf a;
#pragma unroll
        for (int e = 0; e < 8; ++e) { a[e] = alo[e]; a[8 + e] = ahi[e]; }
#pragma unroll
        for (int t = 0; t < 4; ++t) acc[t] = wmma_bf16(a, bcur[t], acc[t]);
    }

#pragma unroll
    for (int t = 0; t < 4; ++t) {
#pragma unroll
        for (int r = 0; r < 8; ++r) {
            int mi = m_tile + r + 8 * hf;     // row in [0,4096)
            int n  = n_base + t * 16 + mn;    // col in [0,1024)
            int bb = mi >> 10, l = mi & 1023;
            int h  = n >> 6,   d = n & 63;
            __bf16 val = (__bf16)acc[t][r];
            if (z < 2) O[((bb * H_ + h) * L_ + l) * DK_ + d] = val;
            else       O[((bb * H_ + h) * DK_ + d) * L_ + l] = val;  // V transposed
        }
    }
}

// ---------------------------------------------------------------------------
// Fused scores + additive mask + masked_fill + softmax.
// Block owns 16 query rows x 1024 keys for one (b,h). 8 waves x 8 n-tiles.
// Scores staged in LDS, softmax'ed there, probs written once to attn output.
// ---------------------------------------------------------------------------
#define SROW 1032  // padded row stride in floats

__global__ void attn_scores_softmax(const __bf16* __restrict__ qh,
                                    const __bf16* __restrict__ kh,
                                    const float*  __restrict__ maskp,
                                    const int*    __restrict__ srcm,
                                    const float*  __restrict__ gammap,
                                    float* __restrict__ attn) {
    extern __shared__ char smemraw[];
    float* s    = (float*)smemraw;     // [16][SROW]
    float* smn  = s + 16 * SROW;       // [1024] src_mask (key dim)
    float* smr  = smn + 1024;          // [16]   src_mask (query rows)
    float* red1 = smr + 16;            // [16][16] max partials
    float* red2 = red1 + 256;          // [16][16] sum partials

    const int mt = blockIdx.x;         // query tile 0..63
    const int h  = blockIdx.y;
    const int b  = blockIdx.z;
    const int m0 = mt * 16;
    const int tid = threadIdx.x;
    const float g = gammap[0];

    for (int j = tid; j < 1024; j += 256) smn[j] = (float)srcm[b * L_ + j];
    if (tid < 16) smr[tid] = (float)srcm[b * L_ + m0 + tid];

    const int wave = tid >> 5, lane = tid & 31;
    const int hf = lane >> 4, mn = lane & 15;

    const __bf16* qrow = qh + ((b * H_ + h) * L_ + (m0 + mn)) * DK_;

    v8f acc[8] = {};
#pragma unroll
    for (int kk = 0; kk < DK_; kk += 32) {
        v8bf alo = *reinterpret_cast<const v8bf*>(qrow + kk + 16 * hf);
        v8bf ahi = *reinterpret_cast<const v8bf*>(qrow + kk + 8 + 16 * hf);
        v16bf a;
#pragma unroll
        for (int e = 0; e < 8; ++e) { a[e] = alo[e]; a[8 + e] = ahi[e]; }
#pragma unroll
        for (int t = 0; t < 8; ++t) {
            int n = wave * 128 + t * 16 + mn;
            const __bf16* krow = kh + ((b * H_ + h) * L_ + n) * DK_;
            v16bf bf = *reinterpret_cast<const v16bf*>(krow + kk + 16 * hf);
            acc[t] = wmma_bf16(a, bf, acc[t]);
        }
    }

    __syncthreads();  // smn/smr ready

    // scale 1/sqrt(dk)=0.125, + mask*gamma, masked_fill(-1e-9), stage to LDS
#pragma unroll
    for (int t = 0; t < 8; ++t) {
        int n = wave * 128 + t * 16 + mn;
#pragma unroll
        for (int r = 0; r < 8; ++r) {
            int mi = r + 8 * hf;
            float val = acc[t][r] * 0.125f +
                        maskp[(b * L_ + (m0 + mi)) * L_ + n] * g;
            if (smr[mi] * smn[n] == 0.0f) val = -1e-9f;
            s[mi * SROW + n] = val;
        }
    }
    __syncthreads();

    // Row softmax: 16 threads per row, 64 elements each.
    const int r = tid >> 4;
    const int c = tid & 15;
    float pmax = -3.4e38f;
    for (int j = c; j < 1024; j += 16) pmax = fmaxf(pmax, s[r * SROW + j]);
    red1[r * 16 + c] = pmax;
    __syncthreads();
    float rmax = -3.4e38f;
#pragma unroll
    for (int i = 0; i < 16; ++i) rmax = fmaxf(rmax, red1[r * 16 + i]);

    float psum = 0.0f;
    for (int j = c; j < 1024; j += 16) {
        float e = __expf(s[r * SROW + j] - rmax);
        s[r * SROW + j] = e;
        psum += e;
    }
    red2[r * 16 + c] = psum;
    __syncthreads();
    float rsum = 0.0f;
#pragma unroll
    for (int i = 0; i < 16; ++i) rsum += red2[r * 16 + i];
    float inv = 1.0f / rsum;

    float* arow = attn + (long)((b * H_ + h) * L_ + (m0 + r)) * L_;
    for (int j = c; j < 1024; j += 16) arow[j] = s[r * SROW + j] * inv;
}

// ---------------------------------------------------------------------------
// out = P @ V.  P: f32 probs (attn region), converted to bf16 fragments
// in-register.  V held transposed [B,H,64,L] so B-fragments are contiguous.
// Wave computes 16(M) x 64(N=dv); K = 1024 key positions. Double-buffered.
// ---------------------------------------------------------------------------
__global__ void pv_kernel(const float* __restrict__ attn,
                          const __bf16* __restrict__ vht,
                          float* __restrict__ out) {
    const int wave = threadIdx.x >> 5, lane = threadIdx.x & 31;
    const int hf = lane >> 4, mn = lane & 15;
    const int h = blockIdx.y, b = blockIdx.z;
    const int m0 = (blockIdx.x * 8 + wave) * 16;

    const float*  prow = attn + (long)((b * H_ + h) * L_ + (m0 + mn)) * L_;
    const __bf16* vcol = vht + ((b * H_ + h) * DK_ + mn) * L_ + 16 * hf;

    v8f acc[4] = {};

    // ---- prologue: kk = 0 fragments
    v4f p0 = *reinterpret_cast<const v4f*>(prow + 16 * hf);
    v4f p1 = *reinterpret_cast<const v4f*>(prow + 16 * hf + 4);
    v4f p2 = *reinterpret_cast<const v4f*>(prow + 8 + 16 * hf);
    v4f p3 = *reinterpret_cast<const v4f*>(prow + 8 + 16 * hf + 4);
    v16bf bcur[4];
#pragma unroll
    for (int t = 0; t < 4; ++t)
        bcur[t] = *reinterpret_cast<const v16bf*>(vcol + t * 16 * L_);

    for (int kk = 0; kk < L_ - 32; kk += 32) {
        const int kn = kk + 32;
        v4f q0 = *reinterpret_cast<const v4f*>(prow + kn + 16 * hf);
        v4f q1 = *reinterpret_cast<const v4f*>(prow + kn + 16 * hf + 4);
        v4f q2 = *reinterpret_cast<const v4f*>(prow + kn + 8 + 16 * hf);
        v4f q3 = *reinterpret_cast<const v4f*>(prow + kn + 8 + 16 * hf + 4);
        v16bf bnxt[4];
#pragma unroll
        for (int t = 0; t < 4; ++t)
            bnxt[t] = *reinterpret_cast<const v16bf*>(vcol + t * 16 * L_ + kn);

        v16bf a;
#pragma unroll
        for (int e = 0; e < 4; ++e) {
            a[e]      = (__bf16)p0[e];
            a[4 + e]  = (__bf16)p1[e];
            a[8 + e]  = (__bf16)p2[e];
            a[12 + e] = (__bf16)p3[e];
        }
#pragma unroll
        for (int t = 0; t < 4; ++t) acc[t] = wmma_bf16(a, bcur[t], acc[t]);

        p0 = q0; p1 = q1; p2 = q2; p3 = q3;
#pragma unroll
        for (int t = 0; t < 4; ++t) bcur[t] = bnxt[t];
    }
    // ---- epilogue
    {
        v16bf a;
#pragma unroll
        for (int e = 0; e < 4; ++e) {
            a[e]      = (__bf16)p0[e];
            a[4 + e]  = (__bf16)p1[e];
            a[8 + e]  = (__bf16)p2[e];
            a[12 + e] = (__bf16)p3[e];
        }
#pragma unroll
        for (int t = 0; t < 4; ++t) acc[t] = wmma_bf16(a, bcur[t], acc[t]);
    }

#pragma unroll
    for (int t = 0; t < 4; ++t) {
#pragma unroll
        for (int r = 0; r < 8; ++r) {
            int l = m0 + r + 8 * hf;
            int d = t * 16 + mn;
            out[(long)(b * L_ + l) * (H_ * DK_) + h * DK_ + d] = acc[t][r];
        }
    }
}

// ---------------------------------------------------------------------------
extern "C" void kernel_launch(void* const* d_in, const int* in_sizes, int n_in,
                              void* d_out, int out_size, void* d_ws, size_t ws_size,
                              hipStream_t stream) {
    const float* q     = (const float*)d_in[0];
    const float* k     = (const float*)d_in[1];
    const float* v     = (const float*)d_in[2];
    const float* mask  = (const float*)d_in[3];
    const int*   srcm  = (const int*)d_in[4];
    const float* Wq    = (const float*)d_in[5];
    const float* Wk    = (const float*)d_in[6];
    const float* Wv    = (const float*)d_in[7];
    const float* gamma = (const float*)d_in[8];

    char* ws = (char*)d_ws;
    size_t off = 0;
    const size_t XN = (size_t)B_ * L_ * D_;   // 4 Mi elems
    const size_t WN = (size_t)D_ * D_;        // 1 Mi elems
    __bf16* qb  = (__bf16*)(ws + off); off += XN * 2;
    __bf16* kb  = (__bf16*)(ws + off); off += XN * 2;
    __bf16* vb  = (__bf16*)(ws + off); off += XN * 2;
    __bf16* wqb = (__bf16*)(ws + off); off += WN * 2;
    __bf16* wkb = (__bf16*)(ws + off); off += WN * 2;
    __bf16* wvb = (__bf16*)(ws + off); off += WN * 2;
    __bf16* qhp = (__bf16*)(ws + off); off += XN * 2;  // [B,H,L,64]
    __bf16* khp = (__bf16*)(ws + off); off += XN * 2;  // [B,H,L,64]
    __bf16* vht = (__bf16*)(ws + off); off += XN * 2;  // [B,H,64,L]

    float* outp = (float*)d_out;                        // [B,L,H*64]
    float* attn = outp + (size_t)B_ * L_ * (H_ * DK_);  // [B,H,L,L]

    cast_f32_bf16<<<2048, 256, 0, stream>>>(q,  qb,  (int)XN);
    cast_f32_bf16<<<2048, 256, 0, stream>>>(k,  kb,  (int)XN);
    cast_f32_bf16<<<2048, 256, 0, stream>>>(v,  vb,  (int)XN);
    cast_f32_bf16<<<1024, 256, 0, stream>>>(Wq, wqb, (int)WN);
    cast_f32_bf16<<<1024, 256, 0, stream>>>(Wk, wkb, (int)WN);
    cast_f32_bf16<<<1024, 256, 0, stream>>>(Wv, wvb, (int)WN);

    proj_kernel<<<dim3(16, 32, 3), 256, 0, stream>>>(qb, kb, vb, wqb, wkb, wvb,
                                                     qhp, khp, vht);

    size_t smem = (size_t)(16 * SROW + 1024 + 16 + 256 + 256) * sizeof(float);
    attn_scores_softmax<<<dim3(64, 16, 4), 256, smem, stream>>>(qhp, khp, mask, srcm,
                                                                gamma, attn);

    pv_kernel<<<dim3(8, 16, 4), 256, 0, stream>>>(attn, vht, outp);
}
